// DynamicGCN_47107201302652
// MI455X (gfx1250) — matrix operand order repaired and checked by reading, not compiled, not run on hardware
//
#include <hip/hip_runtime.h>
#include <math.h>

// ---------------------------------------------------------------------------
// DynamicGCN forward for MI455X (gfx1250, wave32, WMMA bf16 16x16x32)
// ---------------------------------------------------------------------------

#define IN_C     128
#define HID_C    256
#define OUT_C    128
#define N_HEADS  8
#define HEAD_DIM 32

typedef __bf16 bf16_t;
typedef __attribute__((ext_vector_type(16))) __bf16 v16bf;
typedef __attribute__((ext_vector_type(8)))  __bf16 v8bf;
typedef __attribute__((ext_vector_type(8)))  float  v8f;

#define DEV __device__ __forceinline__

// ---- WMMA tile loaders -----------------------------------------------------
// A tile 16x32 (MxK), row-major source, ld = row stride in elements.
// ISA layout: lanes 0-15 -> M=lane, K in {k0+0..7, k0+16..23};
//             lanes 16-31 -> M=lane-16, K in {k0+8..15, k0+24..31}.
DEV v16bf load_tileA(const bf16_t* A, int ld, int row0, int k0) {
  const int lane = threadIdx.x & 31;
  const int row  = row0 + (lane & 15);
  const int kb   = k0 + ((lane & 16) ? 8 : 0);
  const bf16_t* p = A + (size_t)row * ld + kb;
  v8bf lo = *(const v8bf*)(p);        // 16B aligned chunk k..k+7
  v8bf hi = *(const v8bf*)(p + 16);   // 16B aligned chunk k+16..k+23
  v16bf a;
#pragma unroll
  for (int i = 0; i < 8; ++i) { a[i] = lo[i]; a[i + 8] = hi[i]; }
  return a;
}

// B tile 32x16 (KxN) where column n of B is row n of BT (pre-transposed /
// K-matrix). ISA layout: lane = column (mod 16); lanes 0-15 hold K
// {k0+0..7, k0+16..23}, lanes 16-31 hold K {k0+8..15, k0+24..31}.
DEV v16bf load_tileB(const bf16_t* BT, int ld, int col0, int k0) {
  const int lane = threadIdx.x & 31;
  const int col  = col0 + (lane & 15);
  const int kb   = k0 + ((lane & 16) ? 8 : 0);
  const bf16_t* p = BT + (size_t)col * ld + kb;
  v8bf lo = *(const v8bf*)(p);
  v8bf hi = *(const v8bf*)(p + 16);
  v16bf b;
#pragma unroll
  for (int i = 0; i < 8; ++i) { b[i] = lo[i]; b[i + 8] = hi[i]; }
  return b;
}

DEV float wave_sum(float v) {
#pragma unroll
  for (int off = 16; off > 0; off >>= 1) v += __shfl_xor(v, off, 32);
  return v;
}

// ---- elementwise helpers ---------------------------------------------------
__global__ void k_zero(float* p, int n) {
  int i = blockIdx.x * blockDim.x + threadIdx.x;
  if (i < n) p[i] = 0.f;
}

__global__ void k_cast(const float* in, bf16_t* out, int n) {
  int i = blockIdx.x * blockDim.x + threadIdx.x;
  if (i < n) out[i] = (bf16_t)in[i];
}

// W [K x N] row-major -> WT [N x K] bf16
__global__ void k_transpose_cast(const float* W, bf16_t* WT, int K, int N) {
  int i = blockIdx.x * blockDim.x + threadIdx.x;
  if (i < K * N) {
    int k = i / N, n = i - k * N;
    WT[(size_t)n * K + k] = (bf16_t)W[i];
  }
}

// ---- WMMA GEMM: C[MxN] = (A[MxK] @ BT^T + bias) * out_scale, opt relu ------
__global__ void k_gemm_bf16(const bf16_t* __restrict__ A,
                            const bf16_t* __restrict__ BT,
                            const float* __restrict__ bias,
                            float* __restrict__ Cf, bf16_t* __restrict__ Cb,
                            int M, int N, int K, float out_scale, int relu) {
  const int wid = blockIdx.x * (blockDim.x >> 5) + (threadIdx.x >> 5);
  const int tn = N >> 4;
  if (wid >= (M >> 4) * tn) return;
  const int mt = (wid / tn) << 4;
  const int nt = (wid % tn) << 4;
  const int lane = threadIdx.x & 31;

  v8f acc = {0.f, 0.f, 0.f, 0.f, 0.f, 0.f, 0.f, 0.f};
  for (int kk = 0; kk < K; kk += 32) {
    v16bf a = load_tileA(A, K, mt, kk);
    v16bf b = load_tileB(BT, K, nt, kk);
    acc = __builtin_amdgcn_wmma_f32_16x16x32_bf16(false, a, false, b,
                                                  (short)0, acc, false, false);
  }
  const int col   = nt + (lane & 15);
  const float bv  = bias ? bias[col] : 0.f;
  const int rbase = mt + ((lane & 16) ? 8 : 0);
#pragma unroll
  for (int r = 0; r < 8; ++r) {
    float v = (acc[r] + bv) * out_scale;
    if (relu) v = fmaxf(v, 0.f);
    if (Cf) Cf[(size_t)(rbase + r) * N + col] = v;
    if (Cb) Cb[(size_t)(rbase + r) * N + col] = (bf16_t)v;
  }
}

// ---- GCN degree / norm / scatter / finalize --------------------------------
__global__ void k_deg(const int* __restrict__ tgt, const float* __restrict__ ew,
                      float* deg, int E) {
  int e = blockIdx.x * blockDim.x + threadIdx.x;
  if (e < E) atomicAdd(&deg[tgt[e]], ew[e]);
}

__global__ void k_dis(const float* __restrict__ deg, float* dis, int n) {
  int i = blockIdx.x * blockDim.x + threadIdx.x;
  if (i < n) {
    float d = deg[i] + 1.f;  // self-loop weight 1
    dis[i] = (d > 0.f) ? rsqrtf(fmaxf(d, 1e-30f)) : 0.f;
  }
}

__global__ void k_scatter(const float* __restrict__ xw, const int* __restrict__ src,
                          const int* __restrict__ tgt, const float* __restrict__ ew,
                          const float* __restrict__ dis, float* agg, int E, int C) {
  int e = blockIdx.x * (blockDim.x >> 5) + (threadIdx.x >> 5);
  if (e >= E) return;
  const int s = src[e], t = tgt[e];
  const float norm = dis[s] * ew[e] * dis[t];
  const int lane = threadIdx.x & 31;
  const float* xr = xw + (size_t)s * C;
  float* ar = agg + (size_t)t * C;
  for (int c = lane; c < C; c += 32) atomicAdd(&ar[c], xr[c] * norm);
}

__global__ void k_finalize(const float* __restrict__ agg, const float* __restrict__ xw,
                           const float* __restrict__ dis, const float* __restrict__ bias,
                           float* outf, bf16_t* outb, int n, int C, int relu) {
  int i = blockIdx.x * blockDim.x + threadIdx.x;
  if (i >= n * C) return;
  const int row = i / C, col = i - row * C;
  const float dd = dis[row];
  float v = agg[i] + xw[i] * dd * dd + bias[col];  // self-loop term
  if (relu) v = fmaxf(v, 0.f);
  if (outf) outf[i] = v;
  if (outb) outb[i] = (bf16_t)v;
}

// ---- flash-style attention row stats (rowmax/rowsum per head per query) ----
// One wave per (head, 16-row tile). Streams 16x16 score tiles via WMMA,
// keeps per-lane online softmax state, merges across the 16-lane halves.
// qbf is pre-scaled by 1/sqrt(HEAD_DIM).
__global__ void k_rowstats(const bf16_t* __restrict__ qbf, const bf16_t* __restrict__ kbf,
                           float* __restrict__ rowmax, float* __restrict__ rowsum, int n) {
  const int wid = blockIdx.x * (blockDim.x >> 5) + (threadIdx.x >> 5);
  const int tiles_m = n >> 4;
  if (wid >= N_HEADS * tiles_m) return;
  const int head = wid / tiles_m;
  const int mt   = (wid % tiles_m) << 4;
  const int lane = threadIdx.x & 31;

  const v16bf a = load_tileA(qbf, HID_C, mt, head * HEAD_DIM);
  float m[8], s[8];
#pragma unroll
  for (int r = 0; r < 8; ++r) { m[r] = -3.0e38f; s[r] = 0.f; }

  for (int ntile = 0; ntile < n; ntile += 16) {
    v16bf b = load_tileB(kbf, HID_C, ntile, head * HEAD_DIM);
    v8f acc = {0.f, 0.f, 0.f, 0.f, 0.f, 0.f, 0.f, 0.f};
    acc = __builtin_amdgcn_wmma_f32_16x16x32_bf16(false, a, false, b,
                                                  (short)0, acc, false, false);
#pragma unroll
    for (int r = 0; r < 8; ++r) {
      float x  = acc[r];
      float nm = fmaxf(m[r], x);
      s[r] = s[r] * __expf(m[r] - nm) + __expf(x - nm);
      m[r] = nm;
    }
  }
  // merge the 16 lanes of each half-wave (row r lives in lanes 0-15 for
  // M=mt+r and lanes 16-31 for M=mt+8+r)
#pragma unroll
  for (int off = 1; off < 16; off <<= 1) {
#pragma unroll
    for (int r = 0; r < 8; ++r) {
      float om = __shfl_xor(m[r], off, 32);
      float os = __shfl_xor(s[r], off, 32);
      float nm = fmaxf(m[r], om);
      s[r] = s[r] * __expf(m[r] - nm) + os * __expf(om - nm);
      m[r] = nm;
    }
  }
  if ((lane & 15) == 0) {
    const int rbase = mt + ((lane & 16) ? 8 : 0);
#pragma unroll
    for (int r = 0; r < 8; ++r) {
      rowmax[(size_t)head * n + rbase + r] = m[r];
      rowsum[(size_t)head * n + rbase + r] = s[r];
    }
  }
}

// ---- per-node L2 norms of h1 -----------------------------------------------
__global__ void k_norms(const float* __restrict__ h, float* nrm, int n) {
  int node = blockIdx.x * (blockDim.x >> 5) + (threadIdx.x >> 5);
  if (node >= n) return;
  const int lane = threadIdx.x & 31;
  const float* p = h + (size_t)node * HID_C;
  float s = 0.f;
  for (int c = lane; c < HID_C; c += 32) { float v = p[c]; s += v * v; }
  s = wave_sum(s);
  if (lane == 0) nrm[node] = fmaxf(sqrtf(s), 1e-8f);
}

// ---- per-edge dynamic weight: cos(h_s,h_t) * mean_h softmax(q_s.k)_t -------
__global__ void k_edgew(const int* __restrict__ src, const int* __restrict__ tgt,
                        const float* __restrict__ h, const float* __restrict__ nrm,
                        const bf16_t* __restrict__ qbf, const bf16_t* __restrict__ kbf,
                        const float* __restrict__ rowmax, const float* __restrict__ rowsum,
                        float* ew2, int E, int n) {
  int e = blockIdx.x * (blockDim.x >> 5) + (threadIdx.x >> 5);
  if (e >= E) return;
  const int lane = threadIdx.x & 31;
  const int s = src[e], t = tgt[e];
  const float* hs = h + (size_t)s * HID_C;
  const float* ht = h + (size_t)t * HID_C;
  float dot = 0.f;
  for (int c = lane; c < HID_C; c += 32) dot += hs[c] * ht[c];
  dot = wave_sum(dot);
  const float cosv = dot / (nrm[s] * nrm[t]);  // ALPHA = 1

  const bf16_t* qs = qbf + (size_t)s * HID_C;
  const bf16_t* kt = kbf + (size_t)t * HID_C;
  float mean = 0.f;
#pragma unroll
  for (int hd = 0; hd < N_HEADS; ++hd) {
    float v = (float)qs[hd * HEAD_DIM + lane] * (float)kt[hd * HEAD_DIM + lane];
    v = wave_sum(v);  // score already scaled (q pre-scaled by 1/sqrt(d))
    mean += __expf(v - rowmax[(size_t)hd * n + s]) / rowsum[(size_t)hd * n + s];
  }
  mean *= (1.f / N_HEADS);
  if (lane == 0) ew2[e] = cosv * mean;
}

// ---------------------------------------------------------------------------
extern "C" void kernel_launch(void* const* d_in, const int* in_sizes, int n_in,
                              void* d_out, int out_size, void* d_ws, size_t ws_size,
                              hipStream_t stream) {
  const float* x  = (const float*)d_in[0];
  const int*   ei = (const int*)d_in[1];
  const float* ew = (const float*)d_in[2];
  const float* W1 = (const float*)d_in[3];  const float* b1 = (const float*)d_in[4];
  const float* W2 = (const float*)d_in[5];  const float* b2 = (const float*)d_in[6];
  const float* W3 = (const float*)d_in[7];  const float* b3 = (const float*)d_in[8];
  const float* Wq = (const float*)d_in[9];  const float* bq = (const float*)d_in[10];
  const float* Wk = (const float*)d_in[11]; const float* bk = (const float*)d_in[12];

  const int n = in_sizes[0] / IN_C;  // 4096
  const int E = in_sizes[2];         // 131072
  const int* src = ei;
  const int* tgt = ei + E;

  // ---- workspace carve-out (~21 MB) ----
  char* wsp = (char*)d_ws;
  size_t off = 0;
  auto carve = [&](size_t bytes) -> void* {
    void* p = wsp + off;
    off += (bytes + 255) & ~(size_t)255;
    return p;
  };
  bf16_t* xbf  = (bf16_t*)carve((size_t)n * IN_C  * 2);
  bf16_t* w1T  = (bf16_t*)carve((size_t)HID_C * IN_C  * 2);
  bf16_t* w2T  = (bf16_t*)carve((size_t)HID_C * HID_C * 2);
  bf16_t* w3T  = (bf16_t*)carve((size_t)OUT_C * HID_C * 2);
  bf16_t* wqT  = (bf16_t*)carve((size_t)HID_C * HID_C * 2);
  bf16_t* wkT  = (bf16_t*)carve((size_t)HID_C * HID_C * 2);
  float*  xw   = (float*) carve((size_t)n * HID_C * 4);
  float*  agg  = (float*) carve((size_t)n * HID_C * 4);
  float*  h1f  = (float*) carve((size_t)n * HID_C * 4);
  bf16_t* h1bf = (bf16_t*)carve((size_t)n * HID_C * 2);
  bf16_t* h2bf = (bf16_t*)carve((size_t)n * HID_C * 2);
  bf16_t* qbf  = (bf16_t*)carve((size_t)n * HID_C * 2);
  bf16_t* kbf  = (bf16_t*)carve((size_t)n * HID_C * 2);
  float*  rmax = (float*) carve((size_t)N_HEADS * n * 4);
  float*  rsum = (float*) carve((size_t)N_HEADS * n * 4);
  float*  deg  = (float*) carve((size_t)n * 4);
  float*  dis  = (float*) carve((size_t)n * 4);
  float*  nrm  = (float*) carve((size_t)n * 4);
  float*  ew2  = (float*) carve((size_t)E * 4);
  (void)ws_size; (void)n_in; (void)out_size;

  const int T = 256;
  auto cdiv = [](int a, int b) { return (a + b - 1) / b; };
  const float qscale = 0.17677669529663687f;  // 1/sqrt(32)

  // ---- precision casts / weight transposes ----
  k_cast<<<cdiv(n * IN_C, T), T, 0, stream>>>(x, xbf, n * IN_C);
  k_transpose_cast<<<cdiv(IN_C * HID_C, T), T, 0, stream>>>(W1, w1T, IN_C, HID_C);
  k_transpose_cast<<<cdiv(HID_C * HID_C, T), T, 0, stream>>>(W2, w2T, HID_C, HID_C);
  k_transpose_cast<<<cdiv(HID_C * OUT_C, T), T, 0, stream>>>(W3, w3T, HID_C, OUT_C);
  k_transpose_cast<<<cdiv(HID_C * HID_C, T), T, 0, stream>>>(Wq, wqT, HID_C, HID_C);
  k_transpose_cast<<<cdiv(HID_C * HID_C, T), T, 0, stream>>>(Wk, wkT, HID_C, HID_C);

  // ---- conv1: h1 = relu(GCN(x, ew=1, W1, b1)) ----
  k_zero<<<cdiv(n * HID_C, T), T, 0, stream>>>(agg, n * HID_C);
  k_zero<<<cdiv(n, T), T, 0, stream>>>(deg, n);
  {
    int tiles = (n >> 4) * (HID_C >> 4);
    k_gemm_bf16<<<cdiv(tiles, 4), 128, 0, stream>>>(xbf, w1T, nullptr, xw, nullptr,
                                                    n, HID_C, IN_C, 1.f, 0);
  }
  k_deg<<<cdiv(E, T), T, 0, stream>>>(tgt, ew, deg, E);
  k_dis<<<cdiv(n, T), T, 0, stream>>>(deg, dis, n);
  k_scatter<<<cdiv(E, 8), T, 0, stream>>>(xw, src, tgt, ew, dis, agg, E, HID_C);
  k_finalize<<<cdiv(n * HID_C, T), T, 0, stream>>>(agg, xw, dis, b1, h1f, h1bf,
                                                   n, HID_C, 1);

  // ---- attention: q/k projections (q pre-scaled by 1/sqrt(d)) ----
  {
    int tiles = (n >> 4) * (HID_C >> 4);
    k_gemm_bf16<<<cdiv(tiles, 4), 128, 0, stream>>>(h1bf, wqT, bq, nullptr, qbf,
                                                    n, HID_C, HID_C, qscale, 0);
    k_gemm_bf16<<<cdiv(tiles, 4), 128, 0, stream>>>(h1bf, wkT, bk, nullptr, kbf,
                                                    n, HID_C, HID_C, 1.f, 0);
  }
  // flash-style row stats: 8 heads x (n/16) row tiles, 4 waves/block
  k_rowstats<<<cdiv(N_HEADS * (n >> 4), 4), 128, 0, stream>>>(qbf, kbf, rmax, rsum, n);
  k_norms<<<cdiv(n, 8), T, 0, stream>>>(h1f, nrm, n);
  k_edgew<<<cdiv(E, 8), T, 0, stream>>>(src, tgt, h1f, nrm, qbf, kbf, rmax, rsum,
                                        ew2, E, n);

  // ---- conv2: h2 = relu(GCN(h1, ew2, W2, b2)) ----
  k_zero<<<cdiv(n * HID_C, T), T, 0, stream>>>(agg, n * HID_C);
  k_zero<<<cdiv(n, T), T, 0, stream>>>(deg, n);
  {
    int tiles = (n >> 4) * (HID_C >> 4);
    k_gemm_bf16<<<cdiv(tiles, 4), 128, 0, stream>>>(h1bf, w2T, nullptr, xw, nullptr,
                                                    n, HID_C, HID_C, 1.f, 0);
  }
  k_deg<<<cdiv(E, T), T, 0, stream>>>(tgt, ew2, deg, E);
  k_dis<<<cdiv(n, T), T, 0, stream>>>(deg, dis, n);
  k_scatter<<<cdiv(E, 8), T, 0, stream>>>(xw, src, tgt, ew2, dis, agg, E, HID_C);
  k_finalize<<<cdiv(n * HID_C, T), T, 0, stream>>>(agg, xw, dis, b2, nullptr, h2bf,
                                                   n, HID_C, 1);

  // ---- conv3: out = GCN(h2, ew2, W3, b3) (same ew2 -> same deg/dis) ----
  k_zero<<<cdiv(n * OUT_C, T), T, 0, stream>>>(agg, n * OUT_C);
  {
    int tiles = (n >> 4) * (OUT_C >> 4);
    k_gemm_bf16<<<cdiv(tiles, 4), 128, 0, stream>>>(h2bf, w3T, nullptr, xw, nullptr,
                                                    n, OUT_C, HID_C, 1.f, 0);
  }
  k_scatter<<<cdiv(E, 8), T, 0, stream>>>(xw, src, tgt, ew2, dis, agg, E, OUT_C);
  k_finalize<<<cdiv(n * OUT_C, T), T, 0, stream>>>(agg, xw, dis, b3, (float*)d_out,
                                                   nullptr, n, OUT_C, 0);
}
